// xLSTMClassifier_807453851722
// MI455X (gfx1250) — matrix operand order repaired and checked by reading, not compiled, not run on hardware
//
#include <hip/hip_runtime.h>
#include <hip/hip_bf16.h>

// xLSTM forward for MI455X (gfx1250, wave32, WMMA).
// GEMMs + S*S mLSTM parallel part: bf16 WMMA 16x16x32, fp32 accumulate.
// Fragment-packed LDS tiles; staging is fully vectorized (global b128 loads ->
// packed bf16 -> ds b128/b32 stores), no per-element bounds guards (all shapes
// in this model are exact tile multiples: M%256==0, N%64==0, K%32==0).
// sLSTM recurrence: R transposed+padded in LDS, float4 (b128) inner loop.

#define S_LEN  1024
#define DMODEL 512
#define INNER_ 1024
#define NHM_   4
#define DHM_   256
#define NHS_   4
#define DHS_   128

typedef __attribute__((ext_vector_type(16))) __bf16 v16bf;
typedef __attribute__((ext_vector_type(8)))  __bf16 v8bf;
typedef __attribute__((ext_vector_type(8)))  float  v8f;

// ---------- bf16 helpers (round-to-nearest-even) ----------
__device__ __forceinline__ unsigned short f2us(float f) {
  unsigned int u = __float_as_uint(f);
  u += 0x7FFFu + ((u >> 16) & 1u);
  return (unsigned short)(u >> 16);
}
__device__ __forceinline__ unsigned int pack2(float a, float b) {
  return (unsigned int)f2us(a) | ((unsigned int)f2us(b) << 16);
}
__device__ __forceinline__ float logsigf(float x) {
  return fminf(x, 0.f) - log1pf(__expf(-fabsf(x)));
}
__device__ __forceinline__ float siluf(float x) {
  return x / (1.f + __expf(-x));
}

// A-fragment inverse map (used for score->P register scatter only).
// Lane (m16 + 16*half) holds e = region*8 + w8 for k = region*16 + half*8 + w8.
__device__ __forceinline__ int fragA_pos(int m16, int k32) {
  int region = k32 >> 4;
  int within = k32 & 15;
  int half = within >> 3;
  int w8 = within & 7;
  return (m16 + 16 * half) * 16 + region * 8 + w8;
}
// Load one lane's packed 32B fragment chunk as 2x16B LDS loads.
__device__ __forceinline__ v16bf load_frag(const unsigned short* p) {
  const v8bf* q = (const v8bf*)p;
  v8bf lo = q[0], hi = q[1];
  return __builtin_shufflevector(lo, hi, 0, 1, 2, 3, 4, 5, 6, 7, 8, 9, 10, 11,
                                 12, 13, 14, 15);
}
__device__ __forceinline__ v8f wmma_bf16(v16bf a, v16bf b, v8f c) {
  return __builtin_amdgcn_wmma_f32_16x16x32_bf16(false, a, false, b, (short)0, c,
                                                 false, false);
}

// ---------- embedding gather ----------
__global__ __launch_bounds__(256) void k_embed(const int* __restrict__ ids,
                                               const float* __restrict__ emb,
                                               float* __restrict__ h) {
  int row = blockIdx.x;
  int tok = ids[row];
  for (int c = threadIdx.x; c < DMODEL; c += 256)
    h[(size_t)row * DMODEL + c] = emb[(size_t)tok * DMODEL + c];
}

// ---------- rowwise layernorm ----------
__global__ __launch_bounds__(256) void k_layernorm(const float* __restrict__ in, int ldin,
                                                   const float* __restrict__ w,
                                                   float* __restrict__ out, int ldout,
                                                   int W) {
  __shared__ float red[256];
  int row = blockIdx.x, tid = threadIdx.x;
  const float* rp = in + (size_t)row * ldin;
  float s = 0.f;
  for (int c = tid; c < W; c += 256) s += rp[c];
  red[tid] = s; __syncthreads();
  for (int st = 128; st > 0; st >>= 1) { if (tid < st) red[tid] += red[tid + st]; __syncthreads(); }
  float mu = red[0] / (float)W; __syncthreads();
  float s2 = 0.f;
  for (int c = tid; c < W; c += 256) { float d = rp[c] - mu; s2 += d * d; }
  red[tid] = s2; __syncthreads();
  for (int st = 128; st > 0; st >>= 1) { if (tid < st) red[tid] += red[tid + st]; __syncthreads(); }
  float rs = rsqrtf(red[0] / (float)W + 1e-5f);
  for (int c = tid; c < W; c += 256)
    out[(size_t)row * ldout + c] = (rp[c] - mu) * rs * w[c];
}

// ---------- generic WMMA GEMM: C[M,N] = A[M,K] @ B, B[k][n]=Bp[k*bsk+n*bsn]
// optional residual addsrc. BM=256 BN=64 BK=32, 8 waves, 32 rows/wave.
// REQUIRES: M%256==0, N%64==0, K%32==0 (true for every call in this model).
__global__ __launch_bounds__(256) void k_gemm(const float* __restrict__ A, int lda,
                                              const float* __restrict__ B, int bsk, int bsn,
                                              float* __restrict__ C, int ldc,
                                              const float* __restrict__ addsrc, int ldadd,
                                              int M, int N, int K) {
  __shared__ __align__(16) unsigned short As[16 * 512];  // 16 row-subtiles, frag-packed
  __shared__ __align__(16) unsigned short Bs[4 * 512];   // 4 col-subtiles, frag-packed
  int tid = threadIdx.x, lane = tid & 31, wave = tid >> 5;
  int m0 = blockIdx.y * 256, n0 = blockIdx.x * 64;
  v8f acc[8];
#pragma unroll
  for (int i = 0; i < 8; ++i)
    for (int j = 0; j < 8; ++j) acc[i][j] = 0.f;

  for (int k0 = 0; k0 < K; k0 += 32) {
    __syncthreads();
    // A tile 256x32: 1024 chunks of 8 consecutive k -> b128 global + b128 LDS.
#pragma unroll
    for (int i = 0; i < 4; ++i) {
      int grp = i * 256 + tid;  // 0..1023
      int r = grp >> 2;         // row 0..255
      int gk = grp & 3;         // 8-k group
      const float* src = &A[(size_t)(m0 + r) * lda + k0 + gk * 8];
      float4 f0 = ((const float4*)src)[0];
      float4 f1 = ((const float4*)src)[1];
      uint4 pk;
      pk.x = pack2(f0.x, f0.y);
      pk.y = pack2(f0.z, f0.w);
      pk.z = pack2(f1.x, f1.y);
      pk.w = pack2(f1.z, f1.w);
      int lane_d = (r & 15) + 16 * (gk & 1);
      *(uint4*)(As + (r >> 4) * 512 + lane_d * 16 + (gk >> 1) * 8) = pk;
    }
    // B tile 32x64: pack k-pairs -> one b32 LDS store per 2 elements.
#pragma unroll
    for (int i = 0; i < 4; ++i) {
      int pid = i * 256 + tid;   // 0..1023
      int kk = (pid >> 6) << 1;  // even k 0..30
      int c = pid & 63;          // n
      float b0 = B[(size_t)(k0 + kk) * bsk + (size_t)(n0 + c) * bsn];
      float b1 = B[(size_t)(k0 + kk + 1) * bsk + (size_t)(n0 + c) * bsn];
      *(unsigned int*)(Bs + (c >> 4) * 512 + ((c & 15) + 16 * ((kk >> 3) & 1)) * 16 +
                       (kk >> 4) * 8 + (kk & 7)) = pack2(b0, b1);
    }
    if (k0 + 32 < K) {  // gfx1250 global_prefetch of next tiles
      __builtin_prefetch(&A[(size_t)(m0 + tid) * lda + k0 + 32], 0, 1);
      __builtin_prefetch(&B[(size_t)(k0 + 32 + (tid & 31)) * bsk + (size_t)n0 * bsn], 0, 1);
    }
    __syncthreads();
    v16bf af0 = load_frag(As + (wave * 2 + 0) * 512 + lane * 16);
    v16bf af1 = load_frag(As + (wave * 2 + 1) * 512 + lane * 16);
#pragma unroll
    for (int nt = 0; nt < 4; ++nt) {
      v16bf bfrag = load_frag(Bs + nt * 512 + lane * 16);
      acc[nt]     = wmma_bf16(af0, bfrag, acc[nt]);
      acc[4 + nt] = wmma_bf16(af1, bfrag, acc[4 + nt]);
    }
  }
  int nl = lane & 15, half = lane >> 4;
#pragma unroll
  for (int sub = 0; sub < 2; ++sub)
#pragma unroll
    for (int nt = 0; nt < 4; ++nt)
#pragma unroll
      for (int rr = 0; rr < 8; ++rr) {
        int m = m0 + (wave * 2 + sub) * 16 + rr + 8 * half;
        int n = n0 + nt * 16 + nl;
        float vv = acc[sub * 4 + nt][rr];
        if (addsrc) vv += addsrc[(size_t)m * ldadd + n];
        C[(size_t)m * ldc + n] = vv;
      }
}

// ---------- causal depthwise conv (k=3) + SiLU ----------
__global__ __launch_bounds__(256) void k_conv_silu(const float* __restrict__ in, int ldin,
                                                   int coloff, const float* __restrict__ w,
                                                   const float* __restrict__ bias,
                                                   float* __restrict__ out, int ldout, int C) {
  int row = blockIdx.x;
  int b = row / S_LEN, s = row % S_LEN;
  for (int c = threadIdx.x; c < C; c += 256) {
    float acc = bias[c];
#pragma unroll
    for (int i = 0; i < 3; ++i) {
      int sp = s + i - 2;
      if (sp >= 0) acc += in[(size_t)(b * S_LEN + sp) * ldin + coloff + c] * w[c * 3 + i];
    }
    out[(size_t)row * ldout + c] = siluf(acc);
  }
}

// ---------- block-diagonal 4x4 q/k/v projections ----------
__global__ __launch_bounds__(256) void k_blockdiag(const float* __restrict__ xc,
                                                   const float* __restrict__ xin,
                                                   const float* __restrict__ Wq,
                                                   const float* __restrict__ Wk,
                                                   const float* __restrict__ Wv,
                                                   float* __restrict__ q,
                                                   float* __restrict__ k,
                                                   float* __restrict__ v) {
  int row = blockIdx.x, nb = threadIdx.x;  // nb in [0,256)
  float4 xi  = *(const float4*)(xc  + (size_t)row * INNER_ + nb * 4);
  float4 xm4 = *(const float4*)(xin + (size_t)row * (2 * INNER_) + nb * 4);
  const float* wq = Wq + nb * 16;
  const float* wk = Wk + nb * 16;
  const float* wv = Wv + nb * 16;
  float qo[4], ko[4], vo[4];
#pragma unroll
  for (int o = 0; o < 4; ++o) {
    float4 a = *(const float4*)(wq + o * 4);
    float4 b = *(const float4*)(wk + o * 4);
    float4 cw = *(const float4*)(wv + o * 4);
    qo[o] = xi.x * a.x + xi.y * a.y + xi.z * a.z + xi.w * a.w;
    ko[o] = xi.x * b.x + xi.y * b.y + xi.z * b.z + xi.w * b.w;
    vo[o] = xm4.x * cw.x + xm4.y * cw.y + xm4.z * cw.z + xm4.w * cw.w;
  }
  *(float4*)(q + (size_t)row * INNER_ + nb * 4) = make_float4(qo[0], qo[1], qo[2], qo[3]);
  *(float4*)(k + (size_t)row * INNER_ + nb * 4) = make_float4(ko[0], ko[1], ko[2], ko[3]);
  *(float4*)(v + (size_t)row * INNER_ + nb * 4) = make_float4(vo[0], vo[1], vo[2], vo[3]);
}

// ---------- ig/fg gates: (qkv @ Wi + bi), (qkv @ Wf + bf) -> [b][h][s] ----------
__global__ __launch_bounds__(256) void k_gates(const float* __restrict__ q,
                                               const float* __restrict__ k,
                                               const float* __restrict__ v,
                                               const float* __restrict__ Wi,
                                               const float* __restrict__ bi,
                                               const float* __restrict__ Wf,
                                               const float* __restrict__ bfg,
                                               float* __restrict__ ig,
                                               float* __restrict__ fg) {
  __shared__ float rbuf[256];
  int row = blockIdx.x, tid = threadIdx.x;
  int b = row / S_LEN, s = row % S_LEN;
  float acc[8];
#pragma unroll
  for (int o = 0; o < 8; ++o) acc[o] = 0.f;
  for (int c = tid; c < 3 * INNER_; c += 256) {
    float xv = (c < INNER_) ? q[(size_t)row * INNER_ + c]
             : (c < 2 * INNER_) ? k[(size_t)row * INNER_ + c - INNER_]
                                : v[(size_t)row * INNER_ + c - 2 * INNER_];
    float4 wi = *(const float4*)(Wi + (size_t)c * 4);
    float4 wf = *(const float4*)(Wf + (size_t)c * 4);
    acc[0] += xv * wi.x; acc[1] += xv * wi.y; acc[2] += xv * wi.z; acc[3] += xv * wi.w;
    acc[4] += xv * wf.x; acc[5] += xv * wf.y; acc[6] += xv * wf.z; acc[7] += xv * wf.w;
  }
  for (int o = 0; o < 8; ++o) {
    rbuf[tid] = acc[o]; __syncthreads();
    for (int st = 128; st > 0; st >>= 1) { if (tid < st) rbuf[tid] += rbuf[tid + st]; __syncthreads(); }
    if (tid == 0) {
      if (o < 4) ig[(size_t)(b * NHM_ + o) * S_LEN + s] = rbuf[0] + bi[o];
      else       fg[(size_t)(b * NHM_ + (o - 4)) * S_LEN + s] = rbuf[0] + bfg[o - 4];
    }
    __syncthreads();
  }
}

// ---------- per (b,h) decay scans: cs=cumsum(logsig fg), u=ig-cs, m=prefixmax(u) ----------
__global__ __launch_bounds__(1024) void k_decay_scan(const float* __restrict__ ig,
                                                     const float* __restrict__ fg,
                                                     float* __restrict__ cs,
                                                     float* __restrict__ u,
                                                     float* __restrict__ mx) {
  __shared__ float buf[1024];
  int bh = blockIdx.x, t = threadIdx.x;
  float lf = logsigf(fg[(size_t)bh * S_LEN + t]);
  buf[t] = lf; __syncthreads();
  for (int off = 1; off < 1024; off <<= 1) {
    float vv = buf[t];
    float add = (t >= off) ? buf[t - off] : 0.f;
    __syncthreads();
    buf[t] = vv + add;
    __syncthreads();
  }
  float csv = buf[t];
  cs[(size_t)bh * S_LEN + t] = csv;
  float uv = ig[(size_t)bh * S_LEN + t] - csv;
  u[(size_t)bh * S_LEN + t] = uv;
  __syncthreads();
  buf[t] = uv; __syncthreads();
  for (int off = 1; off < 1024; off <<= 1) {
    float vv = buf[t];
    float mv = (t >= off) ? buf[t - off] : -3.4e38f;
    __syncthreads();
    buf[t] = fmaxf(vv, mv);
    __syncthreads();
  }
  mx[(size_t)bh * S_LEN + t] = buf[t];
}

// ---------- tiled mLSTM parallel part (flash-attention style, WMMA) ----------
// grid (row_tile=8, head=4, batch=4). 256 thr, 8 waves, 128-row Q tile, 64-col K/V.
// Fragment-packed LDS: QS [8 rsub][8 ksub][512], KS [4 jsub][8 dsub][512],
// VS [2 ksub][16 nsub][512], PS [8 rsub][2 ksub][512] (PS is wave-private).
#define ATTN_LDS 149248
extern "C" __global__ __launch_bounds__(256)
void k_attention(const float* __restrict__ q, const float* __restrict__ k,
                 const float* __restrict__ v, const float* __restrict__ u_arr,
                 const float* __restrict__ m_arr, const float* __restrict__ cs_arr,
                 float* __restrict__ out) {
  extern __shared__ unsigned char dynsmem[];
  unsigned short* QS = (unsigned short*)__builtin_assume_aligned(dynsmem, 16);
  unsigned short* KS = QS + 8 * 8 * 512;   // 32768
  unsigned short* VS = KS + 4 * 8 * 512;   // 16384
  unsigned short* PS = VS + 2 * 16 * 512;  // 16384
  float* uj     = (float*)(PS + 8 * 2 * 512);  // 64
  float* mrow   = uj + 64;                     // 128
  float* csrow  = mrow + 128;                  // 128
  float* rowsum = csrow + 128;                 // 128

  int it = blockIdx.x, hh = blockIdx.y, b = blockIdx.z;
  int tid = threadIdx.x, lane = tid & 31, wave = tid >> 5;
  int i0 = it * 128;
  size_t qoff = (size_t)b * S_LEN * INNER_ + hh * DHM_;
  int bh = b * NHM_ + hh;

  // Q tile 128x256: 4096 chunks of 8 consecutive d -> b128 global + b128 LDS.
#pragma unroll
  for (int i = 0; i < 16; ++i) {
    int grp = i * 256 + tid;  // 0..4095
    int r = grp >> 5;         // row 0..127
    int gk = grp & 31;        // 8-d group; d0 = gk*8
    const float* src = &q[qoff + (size_t)(i0 + r) * INNER_ + gk * 8];
    float4 f0 = ((const float4*)src)[0];
    float4 f1 = ((const float4*)src)[1];
    uint4 pk;
    pk.x = pack2(f0.x, f0.y);
    pk.y = pack2(f0.z, f0.w);
    pk.z = pack2(f1.x, f1.y);
    pk.w = pack2(f1.z, f1.w);
    int w32 = (gk & 3) * 8;  // d offset within 32-chunk
    int lane_d = (r & 15) + 16 * (w32 >> 4);
    *(uint4*)(QS + ((r >> 4) * 8 + (gk >> 2)) * 512 + lane_d * 16 + ((w32 >> 3) & 1) * 8) = pk;
  }
  if (tid < 128) {
    mrow[tid]   = m_arr[(size_t)bh * S_LEN + i0 + tid];
    csrow[tid]  = cs_arr[(size_t)bh * S_LEN + i0 + tid];
    rowsum[tid] = 0.f;
  }
  v8f acc[16];
#pragma unroll
  for (int i = 0; i < 16; ++i)
    for (int j = 0; j < 8; ++j) acc[i][j] = 0.f;

  int r0 = wave * 16;
  int half = lane >> 4, nl = lane & 15;
  int ntiles = (i0 + 128) / 64;
  for (int jt = 0; jt < ntiles; ++jt) {
    int j0 = jt * 64;
    __syncthreads();  // previous PV / rowsum done; KS/VS reusable
    // K tile 64x256 (B-layout, n=j, k=d): per lane-chunk 16 consecutive d.
#pragma unroll
    for (int i = 0; i < 4; ++i) {
      int grp = i * 256 + tid;  // 0..1023
      int r = grp >> 4;         // j local 0..63
      int g16 = grp & 15;       // d chunk; d0 = g16*16
      const float* src = &k[qoff + (size_t)(j0 + r) * INNER_ + g16 * 16];
      float4 f0 = ((const float4*)src)[0];
      float4 f1 = ((const float4*)src)[1];
      float4 f2 = ((const float4*)src)[2];
      float4 f3 = ((const float4*)src)[3];
      uint4 lo, hi;
      lo.x = pack2(f0.x, f0.y); lo.y = pack2(f0.z, f0.w);
      lo.z = pack2(f1.x, f1.y); lo.w = pack2(f1.z, f1.w);
      hi.x = pack2(f2.x, f2.y); hi.y = pack2(f2.z, f2.w);
      hi.z = pack2(f3.x, f3.y); hi.w = pack2(f3.z, f3.w);
      unsigned short* dst = KS + ((r >> 4) * 8 + (g16 >> 1)) * 512 +
                            ((r & 15) + 16 * (g16 & 1)) * 16;
      ((uint4*)dst)[0] = lo;
      ((uint4*)dst)[1] = hi;
    }
    // V tile 64x256 (B-layout, k=j, n=d): pack j-pairs, 4 d per chunk.
#pragma unroll
    for (int i = 0; i < 8; ++i) {
      int grp = i * 256 + tid;  // 0..2047
      int jp = grp >> 6;        // j pair 0..31
      int dq = grp & 63;        // d quad; d0 = dq*4
      int jj = jp * 2, d0 = dq * 4;
      const float* s0 = &v[qoff + (size_t)(j0 + jj) * INNER_ + d0];
      float4 a0 = *(const float4*)s0;
      float4 a1 = *(const float4*)(s0 + INNER_);
      int vhalf = (jj >> 3) & 1, e = ((jj & 31) >> 4) * 8 + (jj & 7);
      unsigned short* base0 = VS + ((jj >> 5) * 16 + (d0 >> 4)) * 512;
      *(unsigned int*)(base0 + (((d0 + 0) & 15) + 16 * vhalf) * 16 + e) = pack2(a0.x, a1.x);
      *(unsigned int*)(base0 + (((d0 + 1) & 15) + 16 * vhalf) * 16 + e) = pack2(a0.y, a1.y);
      *(unsigned int*)(base0 + (((d0 + 2) & 15) + 16 * vhalf) * 16 + e) = pack2(a0.z, a1.z);
      *(unsigned int*)(base0 + (((d0 + 3) & 15) + 16 * vhalf) * 16 + e) = pack2(a0.w, a1.w);
    }
    if (tid < 64) uj[tid] = u_arr[(size_t)bh * S_LEN + j0 + tid];
    __syncthreads();

    // scores S = Q K^T over dh=256
    v8f sc[4];
#pragma unroll
    for (int c4 = 0; c4 < 4; ++c4)
      for (int j = 0; j < 8; ++j) sc[c4][j] = 0.f;
    for (int dsI = 0; dsI < 8; ++dsI) {
      v16bf aq = load_frag(QS + (wave * 8 + dsI) * 512 + lane * 16);
#pragma unroll
      for (int c4 = 0; c4 < 4; ++c4) {
        v16bf bk = load_frag(KS + (c4 * 8 + dsI) * 512 + lane * 16);
        sc[c4] = wmma_bf16(aq, bk, sc[c4]);
      }
    }
    // P = S * dh^-0.5 * exp(u[j]-m[i]) with causal mask; write wave-private PS,
    // accumulate row sums from f32 registers via lane-group shuffles.
    float psum[8];
#pragma unroll
    for (int rr = 0; rr < 8; ++rr) psum[rr] = 0.f;
#pragma unroll
    for (int c4 = 0; c4 < 4; ++c4)
#pragma unroll
      for (int rr = 0; rr < 8; ++rr) {
        int m16 = rr + 8 * half;
        int rloc = r0 + m16;
        int jloc = c4 * 16 + nl;
        float pv = 0.f;
        if (j0 + jloc <= i0 + rloc)
          pv = sc[c4][rr] * 0.0625f * __expf(uj[jloc] - mrow[rloc]);
        psum[rr] += pv;
        PS[(wave * 2 + (c4 >> 1)) * 512 + fragA_pos(m16, (c4 & 1) * 16 + nl)] = f2us(pv);
      }
#pragma unroll
    for (int rr = 0; rr < 8; ++rr) {
      float s = psum[rr];
      s += __shfl_xor(s, 1, 32);
      s += __shfl_xor(s, 2, 32);
      s += __shfl_xor(s, 4, 32);
      s += __shfl_xor(s, 8, 32);
      if (nl == 0) rowsum[r0 + rr + 8 * half] += s;
    }
    // out += P @ V  (PS written by this wave only -> no barrier needed)
    for (int ksI = 0; ksI < 2; ++ksI) {
      v16bf ap = load_frag(PS + (wave * 2 + ksI) * 512 + lane * 16);
#pragma unroll
      for (int nt = 0; nt < 16; ++nt) {
        v16bf bv = load_frag(VS + (ksI * 16 + nt) * 512 + lane * 16);
        acc[nt] = wmma_bf16(ap, bv, acc[nt]);
      }
    }
  }
  __syncthreads();
  if (tid < 128) {
    float nsv = fmaxf(fabsf(rowsum[tid]), __expf(-(csrow[tid] + mrow[tid])));
    rowsum[tid] = 1.0f / (nsv + 1e-6f);
  }
  __syncthreads();
#pragma unroll
  for (int nt = 0; nt < 16; ++nt)
#pragma unroll
    for (int rr = 0; rr < 8; ++rr) {
      int rloc = r0 + rr + 8 * half;
      int dcol = nt * 16 + nl;
      out[qoff + (size_t)(i0 + rloc) * INNER_ + dcol] = acc[nt][rr] * rowsum[rloc];
    }
}

// ---------- mLSTM epilogue: headnorm + skip + silu(z) gate ----------
__global__ __launch_bounds__(256) void k_mlstm_out(const float* __restrict__ attn,
                                                   const float* __restrict__ onorm,
                                                   const float* __restrict__ skip,
                                                   const float* __restrict__ xc,
                                                   const float* __restrict__ xin,
                                                   float* __restrict__ h2) {
  __shared__ float red[256];
  int row = blockIdx.x, tid = threadIdx.x;
  for (int hh = 0; hh < NHM_; ++hh) {
    float vv = attn[(size_t)row * INNER_ + hh * DHM_ + tid];
    red[tid] = vv; __syncthreads();
    for (int st = 128; st > 0; st >>= 1) { if (tid < st) red[tid] += red[tid + st]; __syncthreads(); }
    float mu = red[0] / 256.f; __syncthreads();
    float d = vv - mu;
    red[tid] = d * d; __syncthreads();
    for (int st = 128; st > 0; st >>= 1) { if (tid < st) red[tid] += red[tid + st]; __syncthreads(); }
    float rs = rsqrtf(red[0] / 256.f + 1e-5f); __syncthreads();
    int c = hh * DHM_ + tid;
    float hv = d * rs * onorm[c] + skip[c] * xc[(size_t)row * INNER_ + c];
    float z = xin[(size_t)row * (2 * INNER_) + INNER_ + c];
    h2[(size_t)row * INNER_ + c] = hv * siluf(z);
  }
}

// ---------- sLSTM serial scan: one WG per (b,h); R fp32 resident in LDS ----------
// R stored transposed+padded: Rs[t*132 + d] = R[h][d][t]; inner loop uses float4.
#define RS_STRIDE 132
#define SCAN_LDS ((512 * RS_STRIDE + 4 * 128 + 512) * 4)
extern "C" __global__ __launch_bounds__(512)
void k_slstm_scan(const float* __restrict__ Wx, const float* __restrict__ R,
                  const float* __restrict__ bias, float* __restrict__ ys) {
  extern __shared__ unsigned char dynsmem[];
  float* Rs  = (float*)__builtin_assume_aligned(dynsmem, 16);  // 512*132
  float* y   = Rs + 512 * RS_STRIDE;  // 128
  float* cst = y + 128;
  float* nst = cst + 128;
  float* mst = nst + 128;
  float* raw = mst + 128;             // 512
  int bh = blockIdx.x;
  int b = bh >> 2, hh = bh & 3;
  int t = threadIdx.x;
  for (int i = t; i < 128 * 512; i += 512) {
    int to = i >> 7, d = i & 127;
    Rs[to * RS_STRIDE + d] = R[(size_t)hh * 65536 + (size_t)d * 512 + to];
  }
  if (t < 128) { y[t] = 0.f; cst[t] = 0.f; nst[t] = 0.f; mst[t] = 0.f; }
  __syncthreads();
  int g = t >> 7, e = t & 127;
  float bval = bias[g * (NHS_ * DHS_) + hh * DHS_ + e];
  const float4* rp = (const float4*)(Rs + t * RS_STRIDE);
  for (int s = 0; s < S_LEN; ++s) {
    float a = Wx[(size_t)(b * S_LEN + s) * 2048 + g * 512 + hh * 128 + e] + bval;
#pragma unroll 8
    for (int d4 = 0; d4 < 32; ++d4) {
      float4 yv = *(const float4*)(y + 4 * d4);  // broadcast, conflict-free
      float4 rv = rp[d4];
      a += yv.x * rv.x + yv.y * rv.y + yv.z * rv.z + yv.w * rv.w;
    }
    raw[t] = a;
    __syncthreads();
    if (t < 128) {
      float ir = raw[t], fr = raw[128 + t], zr = raw[256 + t], orr = raw[384 + t];
      float lfm = mst[t] + logsigf(fr);
      float mn = fmaxf(ir, lfm);
      float igt = __expf(ir - mn), fgt = __expf(lfm - mn);
      float cn = fgt * cst[t] + igt * tanhf(zr);
      float nn = fgt * nst[t] + igt;
      float yn = (1.f / (1.f + __expf(-orr))) * cn / nn;
      cst[t] = cn; nst[t] = nn; mst[t] = mn; y[t] = yn;
      ys[(size_t)(b * S_LEN + s) * DMODEL + hh * 128 + t] = yn;
    }
    __syncthreads();
  }
}

// ---------- sLSTM headnorm (+residual into h) ----------
__global__ __launch_bounds__(512) void k_slstm_hn_add(const float* __restrict__ ys,
                                                      const float* __restrict__ gn,
                                                      float* __restrict__ h) {
  __shared__ float red[512];
  int row = blockIdx.x, t = threadIdx.x;
  int hh = t >> 7, e = t & 127;
  float vv = ys[(size_t)row * DMODEL + t];
  red[t] = vv; __syncthreads();
  for (int st = 64; st > 0; st >>= 1) { if (e < st) red[t] += red[t + st]; __syncthreads(); }
  float mu = red[hh * 128] / 128.f; __syncthreads();
  float d = vv - mu;
  red[t] = d * d; __syncthreads();
  for (int st = 64; st > 0; st >>= 1) { if (e < st) red[t] += red[t + st]; __syncthreads(); }
  float rs = rsqrtf(red[hh * 128] / 128.f + 1e-5f); __syncthreads();
  h[(size_t)row * DMODEL + t] += d * rs * gn[t];
}

// ---------- GEGLU-ish: gelu(g) * up ----------
__global__ __launch_bounds__(256) void k_geglu(const float* __restrict__ u,
                                               float* __restrict__ out) {
  int row = blockIdx.x;
  for (int c = threadIdx.x; c < DMODEL; c += 256) {
    float g = u[(size_t)row * (2 * DMODEL) + c];
    float up = u[(size_t)row * (2 * DMODEL) + DMODEL + c];
    float ge = 0.5f * g * (1.f + erff(g * 0.70710678f));
    out[(size_t)row * DMODEL + c] = ge * up;
  }
}

// ---------- final LN (post_ln) on last token + classifier dot ----------
__global__ __launch_bounds__(256) void k_classify(const float* __restrict__ h,
                                                  const float* __restrict__ w,
                                                  const float* __restrict__ clfW,
                                                  const float* __restrict__ clfb,
                                                  float* __restrict__ out) {
  __shared__ float red[256];
  int b = blockIdx.x, tid = threadIdx.x;
  const float* row = h + ((size_t)b * S_LEN + (S_LEN - 1)) * DMODEL;
  float x0 = row[tid], x1 = row[256 + tid];
  red[tid] = x0 + x1; __syncthreads();
  for (int st = 128; st > 0; st >>= 1) { if (tid < st) red[tid] += red[tid + st]; __syncthreads(); }
  float mu = red[0] / 512.f; __syncthreads();
  red[tid] = (x0 - mu) * (x0 - mu) + (x1 - mu) * (x1 - mu); __syncthreads();
  for (int st = 128; st > 0; st >>= 1) { if (tid < st) red[tid] += red[tid + st]; __syncthreads(); }
  float rs = rsqrtf(red[0] / 512.f + 1e-5f); __syncthreads();
  float c0 = (x0 - mu) * rs * w[tid] * clfW[tid] + (x1 - mu) * rs * w[256 + tid] * clfW[256 + tid];
  red[tid] = c0; __syncthreads();
  for (int st = 128; st > 0; st >>= 1) { if (tid < st) red[tid] += red[tid + st]; __syncthreads(); }
  if (tid == 0) out[b] = red[0] + clfb[0];
}

// ============================= host launcher =============================
extern "C" void kernel_launch(void* const* d_in, const int* in_sizes, int n_in,
                              void* d_out, int out_size, void* d_ws, size_t ws_size,
                              hipStream_t stream) {
  (void)in_sizes; (void)n_in; (void)out_size; (void)ws_size;
  auto F = [&](int i) { return (const float*)d_in[i]; };
  const int* xids = (const int*)d_in[0];
  const float* emb = F(1);

  float* ws = (float*)d_ws;
  size_t off = 0;
  auto alloc = [&](size_t n) { float* p = ws + off; off += n; return p; };
  float* h    = alloc((size_t)4096 * 512);
  float* t1   = alloc((size_t)4096 * 512);
  float* t0   = alloc((size_t)4096 * 2048);
  float* xc   = alloc((size_t)4096 * 1024);
  float* qb   = alloc((size_t)4096 * 1024);
  float* kb   = alloc((size_t)4096 * 1024);
  float* vb   = alloc((size_t)4096 * 1024);
  float* attn = alloc((size_t)4096 * 1024);
  float* h2   = alloc((size_t)4096 * 1024);
  float* ysb  = alloc((size_t)4096 * 512);
  float* igb  = alloc((size_t)16 * 1024);
  float* fgb  = alloc((size_t)16 * 1024);
  float* csb  = alloc((size_t)16 * 1024);
  float* ub   = alloc((size_t)16 * 1024);
  float* mxb  = alloc((size_t)16 * 1024);

  k_embed<<<4096, 256, 0, stream>>>(xids, emb, h);

  auto mlstm = [&](int p) {
    const float *ln = F(p), *Wup = F(p + 1), *convW = F(p + 2), *convb = F(p + 3),
                *Wq = F(p + 4), *Wk = F(p + 5), *Wv = F(p + 6), *Wi = F(p + 7),
                *bi = F(p + 8), *Wf = F(p + 9), *bfg = F(p + 10), *skip = F(p + 11),
                *onorm = F(p + 12), *Wdown = F(p + 13);
    k_layernorm<<<4096, 256, 0, stream>>>(h, 512, ln, t1, 512, 512);
    k_gemm<<<dim3(2048 / 64, 4096 / 256), 256, 0, stream>>>(
        t1, 512, Wup, 2048, 1, t0, 2048, nullptr, 0, 4096, 2048, 512);
    k_conv_silu<<<4096, 256, 0, stream>>>(t0, 2048, 0, convW, convb, xc, 1024, 1024);
    k_blockdiag<<<4096, 256, 0, stream>>>(xc, t0, Wq, Wk, Wv, qb, kb, vb);
    k_gates<<<4096, 256, 0, stream>>>(qb, kb, vb, Wi, bi, Wf, bfg, igb, fgb);
    k_decay_scan<<<16, 1024, 0, stream>>>(igb, fgb, csb, ub, mxb);
    k_attention<<<dim3(8, NHM_, 4), 256, ATTN_LDS, stream>>>(qb, kb, vb, ub, mxb, csb, attn);
    k_mlstm_out<<<4096, 256, 0, stream>>>(attn, onorm, skip, xc, t0, h2);
    k_gemm<<<dim3(512 / 64, 4096 / 256), 256, 0, stream>>>(
        h2, 1024, Wdown, 512, 1, h, 512, h, 512, 4096, 512, 1024);
  };

  mlstm(2);  // m0 params at indices 2..15

  {  // sLSTM block, params 30..42
    const float *ln1 = F(30), *convW = F(31), *convb = F(32);
    const float* Wg[4] = {F(33), F(34), F(35), F(36)};  // Wig, Wfg, Wzg, Wog
    const float *R = F(37), *bias = F(38), *gn = F(39), *ln2 = F(40), *Wu = F(41),
                *Wd = F(42);
    k_layernorm<<<4096, 256, 0, stream>>>(h, 512, ln1, t1, 512, 512);
    k_conv_silu<<<4096, 256, 0, stream>>>(t1, 512, 0, convW, convb, xc, 512, 512);
    // Wx[b,s,gate,head,e]: 16 small WMMA GEMMs (per gate/head), W accessed as (N=o,K=i)
    for (int g = 0; g < 4; ++g) {
      const float* src = (g < 2) ? xc : t1;
      for (int hh = 0; hh < 4; ++hh) {
        k_gemm<<<dim3(2, 16), 256, 0, stream>>>(
            src + hh * 128, 512, Wg[g] + (size_t)hh * 128 * 128, 1, 128,
            t0 + g * 512 + hh * 128, 2048, nullptr, 0, 4096, 128, 128);
      }
    }
    k_slstm_scan<<<16, 512, SCAN_LDS, stream>>>(t0, R, bias, ysb);
    k_slstm_hn_add<<<4096, 512, 0, stream>>>(ysb, gn, h);
    k_layernorm<<<4096, 256, 0, stream>>>(h, 512, ln2, t1, 512, 512);
    k_gemm<<<dim3(1024 / 64, 4096 / 256), 256, 0, stream>>>(
        t1, 512, Wu, 1024, 1, t0, 1024, nullptr, 0, 4096, 1024, 512);
    k_geglu<<<4096, 256, 0, stream>>>(t0, xc);
    k_gemm<<<dim3(512 / 64, 4096 / 256), 256, 0, stream>>>(
        xc, 512, Wd, 512, 1, h, 512, h, 512, 4096, 512, 512);
  }

  mlstm(16);  // m2 params at indices 16..29

  k_classify<<<4, 256, 0, stream>>>(h, F(43), F(44), F(45), (float*)d_out);
}
